// PyG_PointNet2_Alpha_Predictor_1_24945170055430
// MI455X (gfx1250) — matrix operand order repaired and checked by reading, not compiled, not run on hardware
//
#include <hip/hip_runtime.h>
#include <hip/hip_bf16.h>
#include <math.h>

// ---------------------------------------------------------------------------
// PointNet++ (PyG) alpha predictor forward for MI455X / gfx1250.
// GEMMs run on v_wmma_f32_16x16x32_f16 with branch-free tiling:
//  - weights pre-packed per launch into B-fragment order (f16, K zero-padded)
//  - activations padded to K%32==0 in workspace, staged via LDS as f16
//  - each wave computes a 32x16 output tile (2 WMMAs share one B fragment)
// ---------------------------------------------------------------------------

typedef __attribute__((ext_vector_type(16))) _Float16 v16h;
typedef __attribute__((ext_vector_type(8)))  _Float16 v8h;
typedef __attribute__((ext_vector_type(4)))  _Float16 v4h;
typedef __attribute__((ext_vector_type(8)))  float    v8f;
typedef __attribute__((ext_vector_type(4)))  float    f4v;

#define BB 4
#define NN 8192

// ---------------------------------------------------------------------------
// Farthest point sampling: one block per batch. Matches reference ordering.
// ---------------------------------------------------------------------------
#define FPS_BLOCK 256
__global__ __launch_bounds__(FPS_BLOCK) void fps_kernel(
    const float* __restrict__ pos, int npts, int nsample,
    int* __restrict__ outidx, float* __restrict__ mind)
{
  const int b = blockIdx.x;
  const float* p = pos + (size_t)b * npts * 3;
  int* oi = outidx + (size_t)b * nsample;
  float* md = mind + (size_t)b * npts;
  const int tid = threadIdx.x;

  __shared__ float sv[FPS_BLOCK];
  __shared__ int   si[FPS_BLOCK];
  __shared__ float scur[3];

  const float x0 = p[0], y0 = p[1], z0 = p[2];
  for (int i = tid; i < npts; i += FPS_BLOCK) {
    const float dx = p[i*3+0]-x0, dy = p[i*3+1]-y0, dz = p[i*3+2]-z0;
    md[i] = dx*dx + dy*dy + dz*dz;
  }
  if (tid == 0) oi[0] = 0;
  __syncthreads();

  for (int s = 1; s < nsample; ++s) {
    float bv = -1.0f; int bi = 0x7fffffff;
    for (int i = tid; i < npts; i += FPS_BLOCK) {
      const float v = md[i];
      if (v > bv) { bv = v; bi = i; }          // strict > keeps first max
    }
    sv[tid] = bv; si[tid] = bi;
    __syncthreads();
    for (int off = FPS_BLOCK >> 1; off > 0; off >>= 1) {
      if (tid < off) {
        const float ov = sv[tid + off]; const int ox = si[tid + off];
        if (ov > sv[tid] || (ov == sv[tid] && ox < si[tid])) { sv[tid] = ov; si[tid] = ox; }
      }
      __syncthreads();
    }
    if (tid == 0) {
      const int best = si[0];
      oi[s] = best;
      scur[0] = p[best*3+0]; scur[1] = p[best*3+1]; scur[2] = p[best*3+2];
    }
    __syncthreads();
    const float cx = scur[0], cy = scur[1], cz = scur[2];
    for (int i = tid; i < npts; i += FPS_BLOCK) {
      const float dx = p[i*3+0]-cx, dy = p[i*3+1]-cy, dz = p[i*3+2]-cz;
      const float d = dx*dx + dy*dy + dz*dz;
      const float o = md[i];
      md[i] = d < o ? d : o;
    }
    __syncthreads();
  }
}

// ---------------------------------------------------------------------------
// Gather / copy rows per batch. idx==nullptr => identity copy.
// ---------------------------------------------------------------------------
__global__ void gather_rows_kernel(
    const float* __restrict__ src, const int* __restrict__ idx,
    float* __restrict__ dst, int nsrc, int nsel, int ncols,
    int src_stride, int dst_stride, int dst_off)
{
  const int b = blockIdx.y;
  const int t = blockIdx.x * blockDim.x + threadIdx.x;
  if (t >= nsel * ncols) return;
  const int r = t / ncols, c = t - r * ncols;
  const int sr = idx ? idx[(size_t)b * nsel + r] : r;
  dst[((size_t)b * nsel + r) * dst_stride + dst_off + c] =
      src[((size_t)b * nsrc + sr) * src_stride + c];
}

// Zero the K-padding columns of an activation buffer (rows global, not per-batch).
__global__ void zero_cols_kernel(float* __restrict__ dst, int rows, int stride,
                                 int col0, int ncols)
{
  const int t = blockIdx.x * blockDim.x + threadIdx.x;
  if (t >= rows * ncols) return;
  const int r = t / ncols, c = t - r * ncols;
  dst[(size_t)r * stride + col0 + c] = 0.0f;
}

// ---------------------------------------------------------------------------
// Pre-pack fp32 weights [Kin, Nout] into f16 B-fragment order, K padded:
// Wp[((kt*ntiles + nt)*32 + lane)*16 + e] = W[kt*32 + 16*(lane>>4) + e][nt*16 + (lane&15)]
// (contiguous K order, same as the A fragments: lanes 0-15 hold K 0..15, lanes 16-31 hold K 16..31)
// ---------------------------------------------------------------------------
__global__ void pack_w_kernel(const float* __restrict__ W, _Float16* __restrict__ Wp,
                              int Kin, int Kpad, int Nout)
{
  const int t = blockIdx.x * blockDim.x + threadIdx.x;
  const int ntiles = Nout >> 4;
  const int ktiles = Kpad >> 5;
  if (t >= ktiles * ntiles * 32) return;
  const int lane = t & 31;
  const int nt = (t >> 5) % ntiles;
  const int kt = (t >> 5) / ntiles;
  const int half = lane >> 4, ml = lane & 15;
  const int n = nt * 16 + ml;
  _Float16* dst = Wp + (size_t)t * 16;
  #pragma unroll
  for (int e = 0; e < 16; ++e) {
    const int k = kt * 32 + (half << 4) + e;
    dst[e] = (k < Kin) ? (_Float16)W[(size_t)k * Nout + n] : (_Float16)0.0f;
  }
}

// ---------------------------------------------------------------------------
// WMMA MLP layer: C[M,Nout] = act(A[M,Kpad] @ Wp + bias)
// Block = 128 threads (4 waves); wave w owns n-tile blockIdx.y*64 + w*16 and
// a 32-row M strip (two 16x16 WMMA tiles sharing one B fragment).
// Requires M%32==0, Nout%64==0, Kpad%32==0, pad columns of A zeroed.
// ---------------------------------------------------------------------------
__global__ __launch_bounds__(128) void wmma_mlp_kernel(
    const float* __restrict__ A, const _Float16* __restrict__ Wp,
    const float* __restrict__ bias, float* __restrict__ C,
    int M, int Kpad, int Nout, int relu)
{
  __shared__ _Float16 sA[32 * 32];     // 2 KB
  const int tid  = threadIdx.x;
  const int wave = tid >> 5;
  const int lane = tid & 31;
  const int m0   = blockIdx.x * 32;
  const int nblk = blockIdx.y * 64;
  const int ntiles = Nout >> 4;
  const int nt     = (nblk >> 4) + wave;
  const int half   = lane >> 4;
  const int ml     = lane & 15;

  // A staging map: 128 threads x 2 float4 = 256 float4 = 32 rows x 8 groups
  const int r0 = tid >> 3;              // rows 0..15   (i = tid)
  const int r1 = r0 + 16;               // rows 16..31  (i = tid + 128)
  const int cg = (tid & 7) << 2;        // 0,4,...,28
  const float* arow0 = A + (size_t)(m0 + r0) * Kpad + cg;
  const float* arow1 = A + (size_t)(m0 + r1) * Kpad + cg;

  v8f acc0 = {};
  v8f acc1 = {};
  const int ktiles = Kpad >> 5;
  for (int kt = 0; kt < ktiles; ++kt) {
    const int k0 = kt << 5;
    // stage A tile: two float4 global loads + f16 convert + two b64 LDS stores
    const f4v a4 = *(const f4v*)(arow0 + k0);
    const f4v b4 = *(const f4v*)(arow1 + k0);
    v4h h4, g4;
    h4[0] = (_Float16)a4[0]; h4[1] = (_Float16)a4[1];
    h4[2] = (_Float16)a4[2]; h4[3] = (_Float16)a4[3];
    g4[0] = (_Float16)b4[0]; g4[1] = (_Float16)b4[1];
    g4[2] = (_Float16)b4[2]; g4[3] = (_Float16)b4[3];
    *(v4h*)&sA[(r0 << 5) + cg] = h4;
    *(v4h*)&sA[(r1 << 5) + cg] = g4;
    __syncthreads();

    // A fragments: 16 contiguous halves per lane, K = 16*(lane>>4) + e (matches packed B)
    const v8h a0lo = *(const v8h*)&sA[(ml << 5) + (half << 4)];
    const v8h a0hi = *(const v8h*)&sA[(ml << 5) + (half << 4) + 8];
    const v8h a1lo = *(const v8h*)&sA[((16 + ml) << 5) + (half << 4)];
    const v8h a1hi = *(const v8h*)&sA[((16 + ml) << 5) + (half << 4) + 8];
    const v16h af0 = __builtin_shufflevector(a0lo, a0hi,
        0,1,2,3,4,5,6,7,8,9,10,11,12,13,14,15);
    const v16h af1 = __builtin_shufflevector(a1lo, a1hi,
        0,1,2,3,4,5,6,7,8,9,10,11,12,13,14,15);

    // B fragment: 32 contiguous bytes per lane from packed weights (shared by
    // both WMMAs of this wave)
    const v16h bf = *(const v16h*)(Wp + (((size_t)kt * ntiles + nt) * 32 + lane) * 16);

    acc0 = __builtin_amdgcn_wmma_f32_16x16x32_f16(
        false, af0, false, bf, (short)0, acc0, false, false);
    acc1 = __builtin_amdgcn_wmma_f32_16x16x32_f16(
        false, af1, false, bf, (short)0, acc1, false, false);
    __syncthreads();
  }

  const int n  = nblk + (wave << 4) + ml;  // D layout: N = lane&15
  const int mh = half << 3;                // M = j + 8*(lane>>4)
  const float bv = bias[n];
  #pragma unroll
  for (int j = 0; j < 8; ++j) {
    float v0 = acc0[j] + bv;
    float v1 = acc1[j] + bv;
    if (relu) { v0 = fmaxf(v0, 0.0f); v1 = fmaxf(v1, 0.0f); }
    C[(size_t)(m0 + mh + j) * Nout + n]      = v0;
    C[(size_t)(m0 + 16 + mh + j) * Nout + n] = v1;
  }
}

// ---------------------------------------------------------------------------
// 3-NN brute force per target point, normalized inverse-d^2 weights.
// ---------------------------------------------------------------------------
__global__ void knn3_kernel(const float* __restrict__ ps, const float* __restrict__ pt,
                            int Ns, int Nt, int* __restrict__ kidx, float* __restrict__ kw)
{
  const int b = blockIdx.y;
  const int t = blockIdx.x * blockDim.x + threadIdx.x;
  if (t >= Nt) return;
  const float* tp = pt + ((size_t)b * Nt + t) * 3;
  const float tx = tp[0], ty = tp[1], tz = tp[2];
  const float* sp = ps + (size_t)b * Ns * 3;
  float d0 = 3.4e38f, d1 = 3.4e38f, d2 = 3.4e38f;
  int   i0 = 0, i1 = 0, i2 = 0;
  for (int i = 0; i < Ns; ++i) {
    const float dx = sp[i*3+0]-tx, dy = sp[i*3+1]-ty, dz = sp[i*3+2]-tz;
    const float d = dx*dx + dy*dy + dz*dz;
    if (d < d0)      { d2=d1; i2=i1; d1=d0; i1=i0; d0=d; i0=i; }
    else if (d < d1) { d2=d1; i2=i1; d1=d;  i1=i; }
    else if (d < d2) { d2=d;  i2=i; }
  }
  const float w0 = 1.0f / fmaxf(d0, 1e-16f);
  const float w1 = 1.0f / fmaxf(d1, 1e-16f);
  const float w2 = 1.0f / fmaxf(d2, 1e-16f);
  const float ws = w0 + w1 + w2;
  const size_t o = ((size_t)b * Nt + t) * 3;
  kidx[o] = i0; kidx[o+1] = i1; kidx[o+2] = i2;
  kw[o] = w0 / ws; kw[o+1] = w1 / ws; kw[o+2] = w2 / ws;
}

__global__ void interp_kernel(const float* __restrict__ feat, const int* __restrict__ kidx,
                              const float* __restrict__ kw, float* __restrict__ dst,
                              int Ns, int Nt, int C, int dst_stride, int dst_off)
{
  const int b = blockIdx.y;
  const int t = blockIdx.x * blockDim.x + threadIdx.x;
  if (t >= Nt * C) return;
  const int r = t / C, c = t - r * C;
  const size_t o = ((size_t)b * Nt + r) * 3;
  const float* fb = feat + (size_t)b * Ns * C;
  const float v = kw[o+0] * fb[(size_t)kidx[o+0] * C + c]
                + kw[o+1] * fb[(size_t)kidx[o+1] * C + c]
                + kw[o+2] * fb[(size_t)kidx[o+2] * C + c];
  dst[((size_t)b * Nt + r) * dst_stride + dst_off + c] = v;
}

// ---------------------------------------------------------------------------
// Head final layer: [64 -> 1] dot + bias, stable softplus. Out [B,1,N] == [B,N].
// ---------------------------------------------------------------------------
__global__ void head_final_kernel(const float* __restrict__ h, const float* __restrict__ w,
                                  const float* __restrict__ bias, float* __restrict__ out, int Mtot)
{
  const int t = blockIdx.x * blockDim.x + threadIdx.x;
  if (t >= Mtot) return;
  const float* hr = h + (size_t)t * 64;
  float acc = bias[0];
  #pragma unroll
  for (int i = 0; i < 64; ++i) acc += hr[i] * w[i];
  out[t] = fmaxf(acc, 0.0f) + log1pf(expf(-fabsf(acc)));   // softplus
}

// ---------------------------------------------------------------------------
extern "C" void kernel_launch(void* const* d_in, const int* in_sizes, int n_in,
                              void* d_out, int out_size, void* d_ws, size_t ws_size,
                              hipStream_t stream)
{
  (void)in_sizes; (void)n_in; (void)out_size; (void)ws_size;
  const float* x   = (const float*)d_in[0];   // [B*N, 6]
  const float* pos = (const float*)d_in[1];   // [B*N, 3]
  // d_in[2] = batch (implied by layout, unused)
  // params flatten (jax dict keys sorted): fp1, fp2, fp3, head, sa1, sa2, sa3
  int p = 3;
  const float *fp1w0=(const float*)d_in[p++], *fp1b0=(const float*)d_in[p++];
  const float *fp1w1=(const float*)d_in[p++], *fp1b1=(const float*)d_in[p++];
  const float *fp1w2=(const float*)d_in[p++], *fp1b2=(const float*)d_in[p++];
  const float *fp2w0=(const float*)d_in[p++], *fp2b0=(const float*)d_in[p++];
  const float *fp2w1=(const float*)d_in[p++], *fp2b1=(const float*)d_in[p++];
  const float *fp3w0=(const float*)d_in[p++], *fp3b0=(const float*)d_in[p++];
  const float *fp3w1=(const float*)d_in[p++], *fp3b1=(const float*)d_in[p++];
  const float *hdw0 =(const float*)d_in[p++], *hdb0 =(const float*)d_in[p++];
  const float *hdw1 =(const float*)d_in[p++], *hdb1 =(const float*)d_in[p++];
  const float *sa1w0=(const float*)d_in[p++], *sa1b0=(const float*)d_in[p++];
  const float *sa1w1=(const float*)d_in[p++], *sa1b1=(const float*)d_in[p++];
  const float *sa1w2=(const float*)d_in[p++], *sa1b2=(const float*)d_in[p++];
  const float *sa2w0=(const float*)d_in[p++], *sa2b0=(const float*)d_in[p++];
  const float *sa2w1=(const float*)d_in[p++], *sa2b1=(const float*)d_in[p++];
  const float *sa2w2=(const float*)d_in[p++], *sa2b2=(const float*)d_in[p++];
  const float *sa3w0=(const float*)d_in[p++], *sa3b0=(const float*)d_in[p++];
  const float *sa3w1=(const float*)d_in[p++], *sa3b1=(const float*)d_in[p++];
  const float *sa3w2=(const float*)d_in[p++], *sa3b2=(const float*)d_in[p++];

  // workspace bump allocator (256B aligned)
  char* ws = (char*)d_ws;
  size_t off = 0;
  auto alloc = [&](size_t bytes) -> void* {
    void* r = ws + off;
    off = (off + bytes + 255) & ~(size_t)255;
    return r;
  };
  int*   I1   = (int*)  alloc((size_t)BB*2048*4);
  int*   I2   = (int*)  alloc((size_t)BB*512*4);
  int*   I3   = (int*)  alloc((size_t)BB*128*4);
  float* P1   = (float*)alloc((size_t)BB*2048*3*4);
  float* P2   = (float*)alloc((size_t)BB*512*3*4);
  float* P3   = (float*)alloc((size_t)BB*128*3*4);
  float* L1X  = (float*)alloc((size_t)BB*2048*128*4);
  float* L2X  = (float*)alloc((size_t)BB*512*256*4);
  float* L3X  = (float*)alloc((size_t)BB*128*1024*4);
  float* L2F  = (float*)alloc((size_t)BB*512*256*4);
  float* L1F  = (float*)alloc((size_t)BB*2048*128*4);
  float* L0F  = (float*)alloc((size_t)BB*NN*128*4);
  float* HH   = (float*)alloc((size_t)BB*NN*64*4);
  float* MIND = (float*)alloc((size_t)BB*NN*4);
  int*   KIDX = (int*)  alloc((size_t)BB*NN*3*4);
  float* KW   = (float*)alloc((size_t)BB*NN*3*4);
  float* BUF0 = (float*)alloc((size_t)BB*NN*160*4);
  float* BUF1 = (float*)alloc((size_t)BB*NN*160*4);

  const dim3 blk1(256);
  auto g1 = [](int n){ return dim3((unsigned)((n + 255) / 256), BB); };
  const int* NOIDX = nullptr;

  // pre-pack all WMMA-layer weights into f16 B-fragment order (K zero-padded)
  auto pack = [&](const float* W, int Kin, int Kpad, int Nout) -> _Float16* {
    _Float16* wp = (_Float16*)alloc((size_t)Kpad * Nout * sizeof(_Float16));
    const int threads = (Kpad / 32) * (Nout / 16) * 32;
    pack_w_kernel<<<dim3((threads + 255) / 256), blk1, 0, stream>>>(W, wp, Kin, Kpad, Nout);
    return wp;
  };
  _Float16* Psa1w0 = pack(sa1w0,    6,   32,   64);
  _Float16* Psa1w1 = pack(sa1w1,   64,   64,   64);
  _Float16* Psa1w2 = pack(sa1w2,   64,   64,  128);
  _Float16* Psa2w0 = pack(sa2w0,  131,  160,  128);
  _Float16* Psa2w1 = pack(sa2w1,  128,  128,  128);
  _Float16* Psa2w2 = pack(sa2w2,  128,  128,  256);
  _Float16* Psa3w0 = pack(sa3w0,  259,  288,  256);
  _Float16* Psa3w1 = pack(sa3w1,  256,  256,  512);
  _Float16* Psa3w2 = pack(sa3w2,  512,  512, 1024);
  _Float16* Pfp3w0 = pack(fp3w0, 1280, 1280,  256);
  _Float16* Pfp3w1 = pack(fp3w1,  256,  256,  256);
  _Float16* Pfp2w0 = pack(fp2w0,  384,  384,  256);
  _Float16* Pfp2w1 = pack(fp2w1,  256,  256,  128);
  _Float16* Pfp1w0 = pack(fp1w0,  134,  160,  128);
  _Float16* Pfp1w1 = pack(fp1w1,  128,  128,  128);
  _Float16* Pfp1w2 = pack(fp1w2,  128,  128,  128);
  _Float16* Phdw0  = pack(hdw0,   128,  128,   64);

  auto mlp = [&](const float* A, const _Float16* Wp, const float* bias, float* C,
                 int M, int Kpad, int Nout, int relu) {
    wmma_mlp_kernel<<<dim3((unsigned)(M / 32), (unsigned)(Nout / 64)), 128, 0, stream>>>(
        A, Wp, bias, C, M, Kpad, Nout, relu);
  };
  auto zcols = [&](float* dst, int rows, int stride, int c0, int nc) {
    const int tot = rows * nc;
    zero_cols_kernel<<<dim3((tot + 255) / 256), blk1, 0, stream>>>(dst, rows, stride, c0, nc);
  };

  // ---- SA1: FPS(N->2048), MLP [6,64,64,128] on gathered x (K padded to 32) ----
  fps_kernel<<<BB, FPS_BLOCK, 0, stream>>>(pos, NN, 2048, I1, MIND);
  gather_rows_kernel<<<g1(2048*3), blk1, 0, stream>>>(pos, I1, P1, NN, 2048, 3, 3, 3, 0);
  gather_rows_kernel<<<g1(2048*6), blk1, 0, stream>>>(x,   I1, BUF0, NN, 2048, 6, 6, 32, 0);
  zcols(BUF0, BB*2048, 32, 6, 26);
  mlp(BUF0, Psa1w0, sa1b0, BUF1, 8192,  32,  64, 1);
  mlp(BUF1, Psa1w1, sa1b1, BUF0, 8192,  64,  64, 1);
  mlp(BUF0, Psa1w2, sa1b2, L1X,  8192,  64, 128, 1);

  // ---- SA2: FPS(2048->512), MLP [131,128,128,256] on [l1x|p2] (K 131->160) ----
  fps_kernel<<<BB, FPS_BLOCK, 0, stream>>>(P1, 2048, 512, I2, MIND);
  gather_rows_kernel<<<g1(512*3),   blk1, 0, stream>>>(P1,  I2,    P2,   2048, 512,   3,   3,   3,   0);
  gather_rows_kernel<<<g1(512*128), blk1, 0, stream>>>(L1X, I2,    BUF0, 2048, 512, 128, 128, 160,   0);
  gather_rows_kernel<<<g1(512*3),   blk1, 0, stream>>>(P2,  NOIDX, BUF0,  512, 512,   3,   3, 160, 128);
  zcols(BUF0, BB*512, 160, 131, 29);
  mlp(BUF0, Psa2w0, sa2b0, BUF1, 2048, 160, 128, 1);
  mlp(BUF1, Psa2w1, sa2b1, BUF0, 2048, 128, 128, 1);
  mlp(BUF0, Psa2w2, sa2b2, L2X,  2048, 128, 256, 1);

  // ---- SA3: FPS(512->128), MLP [259,256,512,1024] on [l2x|p3] (K 259->288) ----
  fps_kernel<<<BB, FPS_BLOCK, 0, stream>>>(P2, 512, 128, I3, MIND);
  gather_rows_kernel<<<g1(128*3),   blk1, 0, stream>>>(P2,  I3,    P3,    512, 128,   3,   3,   3,   0);
  gather_rows_kernel<<<g1(128*256), blk1, 0, stream>>>(L2X, I3,    BUF0,  512, 128, 256, 256, 288,   0);
  gather_rows_kernel<<<g1(128*3),   blk1, 0, stream>>>(P3,  NOIDX, BUF0,  128, 128,   3,   3, 288, 256);
  zcols(BUF0, BB*128, 288, 259, 29);
  mlp(BUF0, Psa3w0, sa3b0, BUF1, 512, 288,  256, 1);
  mlp(BUF1, Psa3w1, sa3b1, BUF0, 512, 256,  512, 1);
  mlp(BUF0, Psa3w2, sa3b2, L3X,  512, 512, 1024, 1);

  // ---- FP3: interp l3x (128 -> 512), MLP [1280,256,256] on [interp|l2x] ----
  knn3_kernel<<<g1(512), blk1, 0, stream>>>(P3, P2, 128, 512, KIDX, KW);
  interp_kernel<<<g1(512*1024), blk1, 0, stream>>>(L3X, KIDX, KW, BUF0, 128, 512, 1024, 1280, 0);
  gather_rows_kernel<<<g1(512*256), blk1, 0, stream>>>(L2X, NOIDX, BUF0, 512, 512, 256, 256, 1280, 1024);
  mlp(BUF0, Pfp3w0, fp3b0, BUF1, 2048, 1280, 256, 1);
  mlp(BUF1, Pfp3w1, fp3b1, L2F,  2048,  256, 256, 1);

  // ---- FP2: interp l2f (512 -> 2048), MLP [384,256,128] on [interp|l1x] ----
  knn3_kernel<<<g1(2048), blk1, 0, stream>>>(P2, P1, 512, 2048, KIDX, KW);
  interp_kernel<<<g1(2048*256), blk1, 0, stream>>>(L2F, KIDX, KW, BUF0, 512, 2048, 256, 384, 0);
  gather_rows_kernel<<<g1(2048*128), blk1, 0, stream>>>(L1X, NOIDX, BUF0, 2048, 2048, 128, 128, 384, 256);
  mlp(BUF0, Pfp2w0, fp2b0, BUF1, 8192, 384, 256, 1);
  mlp(BUF1, Pfp2w1, fp2b1, L1F,  8192, 256, 128, 1);

  // ---- FP1: interp l1f (2048 -> 8192), MLP [134,128,128,128] (K 134->160) ----
  knn3_kernel<<<g1(NN), blk1, 0, stream>>>(P1, pos, 2048, NN, KIDX, KW);
  interp_kernel<<<g1(NN*128), blk1, 0, stream>>>(L1F, KIDX, KW, BUF0, 2048, NN, 128, 160, 0);
  gather_rows_kernel<<<g1(NN*6), blk1, 0, stream>>>(x, NOIDX, BUF0, NN, NN, 6, 6, 160, 128);
  zcols(BUF0, BB*NN, 160, 134, 26);
  mlp(BUF0, Pfp1w0, fp1b0, BUF1, 32768, 160, 128, 1);
  mlp(BUF1, Pfp1w1, fp1b1, BUF0, 32768, 128, 128, 1);
  mlp(BUF0, Pfp1w2, fp1b2, L0F,  32768, 128, 128, 1);

  // ---- head: [128,64] relu via WMMA, then [64,1] + softplus ----
  mlp(L0F, Phdw0, hdb0, HH, 32768, 128, 64, 1);
  {
    dim3 g((32768 + 255) / 256);
    head_final_kernel<<<g, blk1, 0, stream>>>(HH, hdw1, hdb1, (float*)d_out, 32768);
  }
}
